// KimiLayer_4879082848959
// MI455X (gfx1250) — matrix operand order repaired and checked by reading, not compiled, not run on hardware
//
#include <hip/hip_runtime.h>
#include <hip/hip_bf16.h>

typedef __bf16 bf16;
typedef __attribute__((ext_vector_type(16))) __bf16 bf16x16;
typedef __attribute__((ext_vector_type(8)))  __bf16 bf16x8;
typedef __attribute__((ext_vector_type(8)))  float  f32x8;
typedef __attribute__((ext_vector_type(4)))  unsigned int u32x4;
typedef __attribute__((ext_vector_type(8)))  int i32x8;
typedef __attribute__((ext_vector_type(4)))  int i32x4;

#define T_TOK 4096
#define DM 1024
#define FF 2048
#define NE 8
#define NH 16
#define DH 64
#define SEQ 2048
#define MAXSLOT 8320
#define MAXTILE 520

static __device__ inline bf16x16 join8(bf16x8 lo, bf16x8 hi) {
  bf16x16 r;
#pragma unroll
  for (int i = 0; i < 8; ++i) { r[i] = lo[i]; r[i + 8] = hi[i]; }
  return r;
}

// A fragment, 16x32 bf16, row-major A, tile origin p, stride ld.
// ISA layout: lane(half,idx): row=idx; elem e: K = (e<8 ? e : e+8) + 8*half
static __device__ inline bf16x16 load_a_frag(const bf16* p, int ld, int lane) {
  int half = (lane >> 4) & 1, idx = lane & 15;
  const bf16* r = p + (size_t)idx * ld + 8 * half;
  bf16x8 lo = *(const bf16x8*)r;
  bf16x8 hi = *(const bf16x8*)(r + 16);
  return join8(lo, hi);
}

// B fragment, 32x16 bf16, from transposed storage Bt[N,K] row-major.
// ISA layout: lane(half,idx): col=idx; elem e: K = (e<8 ? e : e+8) + 8*half
static __device__ inline bf16x16 load_b_frag(const bf16* pt, int ld, int lane) {
  int half = (lane >> 4) & 1, idx = lane & 15;
  const bf16* r = pt + (size_t)idx * ld + 8 * half;
  bf16x8 lo = *(const bf16x8*)r;
  bf16x8 hi = *(const bf16x8*)(r + 16);
  return join8(lo, hi);
}

static __device__ inline f32x8 wmma_bf16(bf16x16 a, bf16x16 b, f32x8 c) {
  return __builtin_amdgcn_wmma_f32_16x16x32_bf16(false, a, false, b, (short)0, c,
                                                 false, false);
}

// ---- TDM: async 2D tile load (rows x 32 bf16, row stride ld_elems) to LDS ----
// D# per CDNA5 ISA ch.8: group0 = {count/lds_addr/global_addr/type},
// group1 = {data_size, tensor dims, tile dims, dim0 stride}, groups2/3 = 0 (2D).
// This toolchain's builtin takes 6 args (extra i32x8 descriptor words; zero here).
static __device__ inline void tdm_load_2d(unsigned lds_off, const bf16* gptr,
                                          unsigned rows, unsigned ld_elems) {
  unsigned long long ga = (unsigned long long)(size_t)gptr;
  u32x4 g0;
  g0[0] = 1u;                                            // count=1 (valid, user)
  g0[1] = lds_off;                                       // LDS byte address
  g0[2] = (unsigned)ga;                                  // global addr [31:0]
  g0[3] = (unsigned)((ga >> 32) & 0x01FFFFFFu) | (2u << 30); // addr[56:32]|type=2
  i32x8 g1;
  g1[0] = 1 << 16;                                       // data_size = 1 (2 bytes)
  g1[1] = (int)(32u << 16);                              // tensor_dim0 = 32 (lo16)
  g1[2] = (int)((rows & 0xFFFFu) << 16);                 // dim0 hi | tensor_dim1 lo
  g1[3] = (int)((rows >> 16) | (32u << 16));             // dim1 hi | tile_dim0 = 32
  g1[4] = (int)(rows & 0xFFFFu);                         // tile_dim1 = rows, dim2=0
  g1[5] = (int)ld_elems;                                 // tensor_dim0_stride lo32
  g1[6] = 0;                                             // stride0 hi | stride1 lo
  g1[7] = 0;                                             // stride1 hi
  i32x4 gz = {0, 0, 0, 0};
  i32x8 gz8 = {0, 0, 0, 0, 0, 0, 0, 0};
  __builtin_amdgcn_tensor_load_to_lds(g0, g1, gz, gz, gz8, 0);
}

// ---------------- weight transpose + bf16 convert: W[K,N] -> Wt[N,K] -----------
__global__ __launch_bounds__(256) void transconv_kernel(
    const float* __restrict__ W, bf16* __restrict__ Wt, int K, int N) {
  size_t total = (size_t)K * N;
  size_t i = (size_t)blockIdx.x * 256 + threadIdx.x;
  size_t b = blockIdx.z;
  if (i >= total) return;
  int k = (int)(i / N), n = (int)(i % N);
  Wt[b * total + (size_t)n * K + k] = (bf16)W[b * total + i];
}

// ---------------- RMSNorm (fp32 in, bf16 out), one block per token -------------
__global__ __launch_bounds__(256) void rmsnorm_kernel(
    const float* __restrict__ X, const float* __restrict__ w, bf16* __restrict__ out) {
  __shared__ float red[8];
  int t = blockIdx.x, tid = threadIdx.x;
  const float* x = X + (size_t)t * DM;
  float s = 0.f;
#pragma unroll
  for (int j = 0; j < 4; ++j) { float v = x[tid + 256 * j]; s += v * v; }
#pragma unroll
  for (int o = 16; o >= 1; o >>= 1) s += __shfl_xor(s, o, 32);
  if ((tid & 31) == 0) red[tid >> 5] = s;
  __syncthreads();
  if (tid == 0) {
    float tot = 0.f;
    for (int i = 0; i < 8; ++i) tot += red[i];
    red[0] = rsqrtf(tot / DM + 1e-6f);
  }
  __syncthreads();
  float sc = red[0];
#pragma unroll
  for (int j = 0; j < 4; ++j) {
    int i = tid + 256 * j;
    out[(size_t)t * DM + i] = (bf16)(x[i] * sc * w[i]);
  }
}

// ---------------- TDM-fed WMMA GEMM: C[M,N] = A[M,K] * Bt[N,K]^T ---------------
// Block tile 64x256 (8 waves, each 32x64). A/B tiles double-buffered in LDS,
// streamed by the Tensor Data Mover; compute loop is pure ds_load_b128 + WMMA.
// mode 0: fp32 out (+addsrc); mode 1: bf16 row-major out; mode 2: bf16 transposed
__global__ __launch_bounds__(256) void gemm_kernel(
    const bf16* __restrict__ A, const bf16* __restrict__ Bt,
    int M, int N, int K, const float* __restrict__ addsrc,
    float* __restrict__ Cf, bf16* __restrict__ Cb, int mode, int ldct) {
  __shared__ bf16 bstage[2][256 * 32];
  __shared__ bf16 astage[2][64 * 32];
  int tid = threadIdx.x;
  int wv = tid >> 5, lane = tid & 31;
  int half = (lane >> 4) & 1, idx = lane & 15;
  int mblk = blockIdx.x * 64;
  int nblk = blockIdx.y * 256;
  int mw = (wv >> 2) * 32;
  int nw = (wv & 3) * 64;
  f32x8 acc[2][4] = {};
  int nsteps = K / 32;
  if (wv == 0) {
    tdm_load_2d((unsigned)(size_t)&astage[0][0], A + (size_t)mblk * K, 64, K);
    tdm_load_2d((unsigned)(size_t)&bstage[0][0], Bt + (size_t)nblk * K, 256, K);
  }
  for (int i = 0; i < nsteps; ++i) {
    if (wv == 0) __builtin_amdgcn_s_wait_tensorcnt(0);
    __syncthreads();  // publish buf[i&1]; closes iter i-1 reads of buf[(i+1)&1]
    if (wv == 0 && i + 1 < nsteps) {
      int kn = (i + 1) * 32;
      tdm_load_2d((unsigned)(size_t)&astage[(i + 1) & 1][0],
                  A + (size_t)mblk * K + kn, 64, K);
      tdm_load_2d((unsigned)(size_t)&bstage[(i + 1) & 1][0],
                  Bt + (size_t)nblk * K + kn, 256, K);
    }
    const bf16* ab = &astage[i & 1][0];
    const bf16* bb = &bstage[i & 1][0];
    const bf16* r0 = ab + (mw + idx) * 32 + 8 * half;
    bf16x16 a0 = join8(*(const bf16x8*)r0, *(const bf16x8*)(r0 + 16));
    const bf16* r1 = ab + (mw + 16 + idx) * 32 + 8 * half;
    bf16x16 a1 = join8(*(const bf16x8*)r1, *(const bf16x8*)(r1 + 16));
#pragma unroll
    for (int j = 0; j < 4; ++j) {
      const bf16* rb = bb + (nw + 16 * j + idx) * 32 + 8 * half;
      bf16x16 b = join8(*(const bf16x8*)rb, *(const bf16x8*)(rb + 16));
      acc[0][j] = wmma_bf16(a0, b, acc[0][j]);
      acc[1][j] = wmma_bf16(a1, b, acc[1][j]);
    }
  }
  int m0 = mblk + mw;
  int n0 = nblk + nw;
#pragma unroll
  for (int i = 0; i < 2; ++i)
#pragma unroll
    for (int j = 0; j < 4; ++j)
#pragma unroll
      for (int r = 0; r < 8; ++r) {
        int m = m0 + 16 * i + r + 8 * half;
        int n = n0 + 16 * j + idx;
        float v = acc[i][j][r];
        if (mode == 0) {
          if (addsrc) v += addsrc[(size_t)m * N + n];
          Cf[(size_t)m * N + n] = v;
        } else if (mode == 1) {
          Cb[(size_t)m * N + n] = (bf16)v;
        } else {
          Cb[(size_t)n * ldct + m] = (bf16)v;
        }
      }
}

// ---------------- flash attention: one wave per (batch, head, 16-query tile) ---
__global__ __launch_bounds__(256) void attn_kernel(
    const bf16* __restrict__ Q, const bf16* __restrict__ Kb,
    const bf16* __restrict__ Vt, bf16* __restrict__ ctx) {
  __shared__ bf16 pls[8][16 * 32];
  int tid = threadIdx.x;
  int wv = tid >> 5, lane = tid & 31;
  int half = (lane >> 4) & 1, idx = lane & 15;
  int wg = blockIdx.x * 8 + wv;
  int mtile = wg & 127, head = (wg >> 7) & 15, batch = wg >> 11;
  int t0 = batch * SEQ + mtile * 16;
  int kcol = head * DH;
  bf16x16 aq0 = load_a_frag(Q + (size_t)t0 * DM + kcol, DM, lane);
  bf16x16 aq1 = load_a_frag(Q + (size_t)t0 * DM + kcol + 32, DM, lane);
  f32x8 accO[4] = {};
  float mrow[8], lrow[8];
#pragma unroll
  for (int r = 0; r < 8; ++r) { mrow[r] = -1e30f; lrow[r] = 0.f; }
  for (int jt = 0; jt < SEQ / 32; ++jt) {
    int tk = batch * SEQ + jt * 32;
    f32x8 sc[2];
#pragma unroll
    for (int p2 = 0; p2 < 2; ++p2) {
      f32x8 c = {};
      bf16x16 b0 = load_b_frag(Kb + (size_t)(tk + 16 * p2) * DM + kcol, DM, lane);
      c = wmma_bf16(aq0, b0, c);
      bf16x16 b1 = load_b_frag(Kb + (size_t)(tk + 16 * p2) * DM + kcol + 32, DM, lane);
      c = wmma_bf16(aq1, b1, c);
      sc[p2] = c;
    }
    float alpha[8];
#pragma unroll
    for (int r = 0; r < 8; ++r) {
      float s0 = sc[0][r] * 0.125f, s1 = sc[1][r] * 0.125f;
      float rm = fmaxf(s0, s1);
#pragma unroll
      for (int o = 8; o >= 1; o >>= 1) rm = fmaxf(rm, __shfl_xor(rm, o, 32));
      float mn = fmaxf(mrow[r], rm);
      float al = __expf(mrow[r] - mn);
      float p0 = __expf(s0 - mn), p1 = __expf(s1 - mn);
      float rs = p0 + p1;
#pragma unroll
      for (int o = 8; o >= 1; o >>= 1) rs += __shfl_xor(rs, o, 32);
      lrow[r] = lrow[r] * al + rs;
      mrow[r] = mn;
      alpha[r] = al;
      pls[wv][(r + 8 * half) * 32 + idx] = (bf16)p0;
      pls[wv][(r + 8 * half) * 32 + 16 + idx] = (bf16)p1;
    }
#pragma unroll
    for (int j = 0; j < 4; ++j)
#pragma unroll
      for (int r = 0; r < 8; ++r) accO[j][r] *= alpha[r];
    __syncthreads();
    // reload P in A-fragment layout from LDS
    bf16x8 plo = *(const bf16x8*)&pls[wv][idx * 32 + 8 * half];
    bf16x8 phi = *(const bf16x8*)&pls[wv][idx * 32 + 16 + 8 * half];
    bf16x16 ap = join8(plo, phi);
#pragma unroll
    for (int j = 0; j < 4; ++j) {
      bf16x16 bv = load_b_frag(Vt + (size_t)(kcol + 16 * j) * T_TOK + tk, T_TOK, lane);
      accO[j] = wmma_bf16(ap, bv, accO[j]);
    }
    __syncthreads();
  }
  float inv[8];
#pragma unroll
  for (int r = 0; r < 8; ++r) inv[r] = 1.0f / lrow[r];
#pragma unroll
  for (int j = 0; j < 4; ++j)
#pragma unroll
    for (int r = 0; r < 8; ++r)
      ctx[(size_t)(t0 + r + 8 * half) * DM + kcol + 16 * j + idx] =
          (bf16)(accO[j][r] * inv[r]);
}

// ---------------- MoE routing ------------------------------------------------
__global__ void zero_kernel(int* counts, int* cursors) {
  int i = threadIdx.x;
  if (i < NE) { counts[i] = 0; cursors[i] = 0; }
}

__global__ __launch_bounds__(128) void router_kernel(
    const bf16* __restrict__ h2, const float* __restrict__ rw,
    int* __restrict__ topi, float* __restrict__ topw, int* __restrict__ counts) {
  int t = blockIdx.x * 128 + threadIdx.x;
  if (t >= T_TOK) return;
  float acc[NE] = {};
  const bf16* h = h2 + (size_t)t * DM;
  for (int i = 0; i < DM; ++i) {
    float hv = (float)h[i];
#pragma unroll
    for (int e = 0; e < NE; ++e) acc[e] += hv * rw[i * NE + e];
  }
  float m = acc[0];
#pragma unroll
  for (int e = 1; e < NE; ++e) m = fmaxf(m, acc[e]);
  float p[NE], sum = 0.f;
#pragma unroll
  for (int e = 0; e < NE; ++e) { p[e] = __expf(acc[e] - m); sum += p[e]; }
  float invs = 1.f / sum;
#pragma unroll
  for (int e = 0; e < NE; ++e) p[e] *= invs;
  int e0 = 0; float p0 = p[0];
#pragma unroll
  for (int e = 1; e < NE; ++e) if (p[e] > p0) { p0 = p[e]; e0 = e; }
  int e1 = (e0 == 0) ? 1 : 0; float p1 = p[e1];
#pragma unroll
  for (int e = 0; e < NE; ++e) if (e != e0 && p[e] > p1) { p1 = p[e]; e1 = e; }
  float a = __expf(p0), b = __expf(p1);  // reference softmaxes the probabilities
  float w0 = a / (a + b), w1 = b / (a + b);
  topi[t * 2] = e0; topi[t * 2 + 1] = e1;
  topw[t * 2] = w0; topw[t * 2 + 1] = w1;
  atomicAdd(&counts[e0], 1);
  atomicAdd(&counts[e1], 1);
}

__global__ void scan_kernel(const int* counts, int* offsets) {
  if (threadIdx.x == 0 && blockIdx.x == 0) {
    int o = 0;
    for (int e = 0; e < NE; ++e) { offsets[e] = o; o += (counts[e] + 15) & ~15; }
    offsets[NE] = o;
  }
}

__global__ __launch_bounds__(256) void assign_kernel(
    const int* __restrict__ topi, const float* __restrict__ topw,
    const int* __restrict__ offsets, int* __restrict__ cursors,
    int* __restrict__ slot_token, float* __restrict__ slot_w,
    int* __restrict__ tok_slot) {
  int t = blockIdx.x * 256 + threadIdx.x;
  if (t >= T_TOK) return;
#pragma unroll
  for (int k = 0; k < 2; ++k) {
    int e = topi[t * 2 + k];
    int pos = atomicAdd(&cursors[e], 1);
    int slot = offsets[e] + pos;
    slot_token[slot] = t;
    slot_w[slot] = topw[t * 2 + k];
    tok_slot[t * 2 + k] = slot;
  }
}

__global__ void pad_kernel(const int* counts, const int* offsets,
                           int* slot_token, float* slot_w) {
  int e = blockIdx.x;
  int s = offsets[e] + counts[e] + threadIdx.x;
  if (s < offsets[e + 1]) { slot_token[s] = 0; slot_w[s] = 0.f; }
}

// ------- fused gate/up GEMM + SiLU (shared mode or gathered expert mode) ------
__global__ __launch_bounds__(256) void ffn_in_kernel(
    const bf16* __restrict__ A, const bf16* __restrict__ Bg,
    const bf16* __restrict__ Bu, bf16* __restrict__ act,
    const int* __restrict__ slot_token, const float* __restrict__ slot_w,
    const int* __restrict__ offsets) {
  int tile = blockIdx.x;
  int rowbase = tile * 16;
  const bf16 *bg = Bg, *bu = Bu;
  if (offsets) {
    if (rowbase >= offsets[NE]) return;
    int e = 0;
    while (e < NE - 1 && rowbase >= offsets[e + 1]) ++e;
    bg += (size_t)e * FF * DM;
    bu += (size_t)e * FF * DM;
  }
  int tid = threadIdx.x;
  int wv = tid >> 5, lane = tid & 31;
  int half = (lane >> 4) & 1, idx = lane & 15;
  int col0 = blockIdx.y * 512 + wv * 64;
  int tok = slot_token ? slot_token[rowbase + idx] : (rowbase + idx);
  const bf16* arow = A + (size_t)tok * DM + 8 * half;
  f32x8 g[4] = {}, u[4] = {};
  for (int kk = 0; kk < DM; kk += 32) {
    bf16x8 lo = *(const bf16x8*)(arow + kk);
    bf16x8 hi = *(const bf16x8*)(arow + kk + 16);
    bf16x16 a = join8(lo, hi);
#pragma unroll
    for (int j = 0; j < 4; ++j) {
      bf16x16 wg = load_b_frag(bg + (size_t)(col0 + 16 * j) * DM + kk, DM, lane);
      g[j] = wmma_bf16(a, wg, g[j]);
      bf16x16 wu = load_b_frag(bu + (size_t)(col0 + 16 * j) * DM + kk, DM, lane);
      u[j] = wmma_bf16(a, wu, u[j]);
    }
  }
  float wr[8];
#pragma unroll
  for (int r = 0; r < 8; ++r)
    wr[r] = slot_w ? slot_w[rowbase + r + 8 * half] : 1.0f;
#pragma unroll
  for (int j = 0; j < 4; ++j)
#pragma unroll
    for (int r = 0; r < 8; ++r) {
      float gv = g[j][r];
      float av = (gv / (1.f + __expf(-gv))) * u[j][r] * wr[r];
      act[(size_t)(rowbase + r + 8 * half) * FF + col0 + 16 * j + idx] = (bf16)av;
    }
}

// ------- grouped expert down GEMM into per-slot fp32 buffer -------------------
__global__ __launch_bounds__(256) void expert_down_kernel(
    const bf16* __restrict__ gact, const bf16* __restrict__ edt,
    float* __restrict__ specslot, const int* __restrict__ offsets) {
  int base = blockIdx.x * 16;
  if (base >= offsets[NE]) return;
  int e = 0;
  while (e < NE - 1 && base >= offsets[e + 1]) ++e;
  const bf16* B = edt + (size_t)e * DM * FF;
  int tid = threadIdx.x;
  int wv = tid >> 5, lane = tid & 31;
  int half = (lane >> 4) & 1, idx = lane & 15;
  int col0 = blockIdx.y * 512 + wv * 64;
  f32x8 acc[4] = {};
  for (int kk = 0; kk < FF; kk += 32) {
    bf16x16 a = load_a_frag(gact + (size_t)base * FF + kk, FF, lane);
#pragma unroll
    for (int j = 0; j < 4; ++j) {
      bf16x16 b = load_b_frag(B + (size_t)(col0 + 16 * j) * FF + kk, FF, lane);
      acc[j] = wmma_bf16(a, b, acc[j]);
    }
  }
#pragma unroll
  for (int j = 0; j < 4; ++j)
#pragma unroll
    for (int r = 0; r < 8; ++r)
      specslot[(size_t)(base + r + 8 * half) * DM + col0 + 16 * j + idx] = acc[j][r];
}

// ------- deterministic final add: out += spec(slot0) + spec(slot1) ------------
__global__ __launch_bounds__(256) void final_add_kernel(
    float* __restrict__ out, const float* __restrict__ specslot,
    const int* __restrict__ tok_slot) {
  int i = blockIdx.x * 256 + threadIdx.x;
  if (i >= T_TOK * DM) return;
  int t = i >> 10, c = i & (DM - 1);
  int s0 = tok_slot[t * 2], s1 = tok_slot[t * 2 + 1];
  out[i] += specslot[(size_t)s0 * DM + c] + specslot[(size_t)s1 * DM + c];
}

extern "C" void kernel_launch(void* const* d_in, const int* in_sizes, int n_in,
                              void* d_out, int out_size, void* d_ws, size_t ws_size,
                              hipStream_t stream) {
  (void)in_sizes; (void)n_in; (void)out_size; (void)ws_size;
  const float* X   = (const float*)d_in[0];
  const float* n1w = (const float*)d_in[1];
  const float* n2w = (const float*)d_in[2];
  const float* wq  = (const float*)d_in[3];
  const float* wk  = (const float*)d_in[4];
  const float* wvw = (const float*)d_in[5];
  const float* wo  = (const float*)d_in[6];
  const float* rw  = (const float*)d_in[7];
  const float* shg = (const float*)d_in[8];
  const float* shu = (const float*)d_in[9];
  const float* shd = (const float*)d_in[10];
  const float* eg  = (const float*)d_in[11];
  const float* eu  = (const float*)d_in[12];
  const float* ed  = (const float*)d_in[13];
  float* out = (float*)d_out;

  char* ws = (char*)d_ws;
  size_t off = 0;
  auto alloc = [&](size_t bytes) -> void* {
    void* p = ws + off;
    off = (off + bytes + 255) & ~(size_t)255;
    return p;
  };
  bf16* wqt  = (bf16*)alloc((size_t)DM * DM * 2);
  bf16* wkt  = (bf16*)alloc((size_t)DM * DM * 2);
  bf16* wvt  = (bf16*)alloc((size_t)DM * DM * 2);
  bf16* wot  = (bf16*)alloc((size_t)DM * DM * 2);
  bf16* shgt = (bf16*)alloc((size_t)FF * DM * 2);
  bf16* shut = (bf16*)alloc((size_t)FF * DM * 2);
  bf16* shdt = (bf16*)alloc((size_t)DM * FF * 2);
  bf16* egt  = (bf16*)alloc((size_t)NE * FF * DM * 2);
  bf16* eut  = (bf16*)alloc((size_t)NE * FF * DM * 2);
  bf16* edt  = (bf16*)alloc((size_t)NE * DM * FF * 2);
  bf16* hbf  = (bf16*)alloc((size_t)T_TOK * DM * 2);
  bf16* qb   = (bf16*)alloc((size_t)T_TOK * DM * 2);
  bf16* kb   = (bf16*)alloc((size_t)T_TOK * DM * 2);
  bf16* vtb  = (bf16*)alloc((size_t)DM * T_TOK * 2);
  bf16* ctxb = (bf16*)alloc((size_t)T_TOK * DM * 2);
  float* X1  = (float*)alloc((size_t)T_TOK * DM * 4);
  bf16* h2bf = (bf16*)alloc((size_t)T_TOK * DM * 2);
  bf16* actsh = (bf16*)alloc((size_t)T_TOK * FF * 2);
  bf16* gact  = (bf16*)alloc((size_t)MAXSLOT * FF * 2);
  float* specslot = (float*)alloc((size_t)MAXSLOT * DM * 4);
  int* counts   = (int*)alloc(NE * 4);
  int* cursors  = (int*)alloc(NE * 4);
  int* offsets  = (int*)alloc((NE + 1) * 4);
  int* topi     = (int*)alloc((size_t)T_TOK * 2 * 4);
  float* topw   = (float*)alloc((size_t)T_TOK * 2 * 4);
  int* slot_token = (int*)alloc(MAXSLOT * 4);
  float* slot_w   = (float*)alloc(MAXSLOT * 4);
  int* tok_slot   = (int*)alloc((size_t)T_TOK * 2 * 4);

  zero_kernel<<<1, 32, 0, stream>>>(counts, cursors);

  // weight conversion + transpose to bf16 [N,K]
  transconv_kernel<<<dim3((DM * DM + 255) / 256, 1, 1), 256, 0, stream>>>(wq, wqt, DM, DM);
  transconv_kernel<<<dim3((DM * DM + 255) / 256, 1, 1), 256, 0, stream>>>(wk, wkt, DM, DM);
  transconv_kernel<<<dim3((DM * DM + 255) / 256, 1, 1), 256, 0, stream>>>(wvw, wvt, DM, DM);
  transconv_kernel<<<dim3((DM * DM + 255) / 256, 1, 1), 256, 0, stream>>>(wo, wot, DM, DM);
  transconv_kernel<<<dim3((DM * FF + 255) / 256, 1, 1), 256, 0, stream>>>(shg, shgt, DM, FF);
  transconv_kernel<<<dim3((DM * FF + 255) / 256, 1, 1), 256, 0, stream>>>(shu, shut, DM, FF);
  transconv_kernel<<<dim3((FF * DM + 255) / 256, 1, 1), 256, 0, stream>>>(shd, shdt, FF, DM);
  transconv_kernel<<<dim3((DM * FF + 255) / 256, 1, NE), 256, 0, stream>>>(eg, egt, DM, FF);
  transconv_kernel<<<dim3((DM * FF + 255) / 256, 1, NE), 256, 0, stream>>>(eu, eut, DM, FF);
  transconv_kernel<<<dim3((FF * DM + 255) / 256, 1, NE), 256, 0, stream>>>(ed, edt, FF, DM);

  // attention sublayer
  rmsnorm_kernel<<<T_TOK, 256, 0, stream>>>(X, n1w, hbf);
  dim3 gqkv(T_TOK / 64, DM / 256);
  gemm_kernel<<<gqkv, 256, 0, stream>>>(hbf, wqt, T_TOK, DM, DM, nullptr, nullptr, qb, 1, 0);
  gemm_kernel<<<gqkv, 256, 0, stream>>>(hbf, wkt, T_TOK, DM, DM, nullptr, nullptr, kb, 1, 0);
  gemm_kernel<<<gqkv, 256, 0, stream>>>(hbf, wvt, T_TOK, DM, DM, nullptr, nullptr, vtb, 2, T_TOK);
  attn_kernel<<<512, 256, 0, stream>>>(qb, kb, vtb, ctxb);
  gemm_kernel<<<gqkv, 256, 0, stream>>>(ctxb, wot, T_TOK, DM, DM, X, X1, nullptr, 0, 0);

  // MoE sublayer
  rmsnorm_kernel<<<T_TOK, 256, 0, stream>>>(X1, n2w, h2bf);
  router_kernel<<<T_TOK / 128, 128, 0, stream>>>(h2bf, rw, topi, topw, counts);
  scan_kernel<<<1, 1, 0, stream>>>(counts, offsets);
  assign_kernel<<<T_TOK / 256, 256, 0, stream>>>(topi, topw, offsets, cursors,
                                                 slot_token, slot_w, tok_slot);
  pad_kernel<<<NE, 16, 0, stream>>>(counts, offsets, slot_token, slot_w);

  // shared FFN
  ffn_in_kernel<<<dim3(T_TOK / 16, FF / 512), 256, 0, stream>>>(
      h2bf, shgt, shut, actsh, nullptr, nullptr, nullptr);
  gemm_kernel<<<dim3(T_TOK / 64, DM / 256), 256, 0, stream>>>(
      actsh, shdt, T_TOK, DM, FF, X1, out, nullptr, 0, 0);

  // expert FFN (sparse, grouped by expert)
  ffn_in_kernel<<<dim3(MAXTILE, FF / 512), 256, 0, stream>>>(
      h2bf, egt, eut, gact, slot_token, slot_w, offsets);
  expert_down_kernel<<<dim3(MAXTILE, DM / 512), 256, 0, stream>>>(
      gact, edt, specslot, offsets);
  final_add_kernel<<<(T_TOK * DM) / 256, 256, 0, stream>>>(out, specslot, tok_slot);
}
